// SimplifiedAttn_33913061769232
// MI455X (gfx1250) — matrix operand order repaired and checked by reading, not compiled, not run on hardware
//
#include <hip/hip_runtime.h>

// ---------------------------------------------------------------------------
// MI455X (gfx1250) fused attention + LayerNorm, flash style.
//   pass 1: q,k -> f16 (row major), v -> f16 transposed [b][d][n]  (in d_ws)
//   pass 2: per 16-row query tile per wave: online-softmax flash attention
//           with v_wmma_f32_16x16x32_f16, double-buffered ASYNC global->LDS
//           staging (GLOBAL_LOAD_ASYNC_TO_LDS_B128 / ASYNCcnt), then
//           residual + LayerNorm epilogue in fp32.
// ---------------------------------------------------------------------------

typedef _Float16 f16;
typedef __attribute__((ext_vector_type(4)))  _Float16 v4h;
typedef __attribute__((ext_vector_type(8)))  _Float16 v8h;
typedef __attribute__((ext_vector_type(16))) _Float16 v16h;
typedef __attribute__((ext_vector_type(8)))  float    v8f;
typedef __attribute__((ext_vector_type(4)))  float    v4f;
typedef __attribute__((ext_vector_type(4)))  int      v4i;

#define DIMD     256
#define NSEQ     4096
#define NBATCH   8
#define NW       4          // waves per block
#define BM       16         // query rows per wave
#define BLOCK_M  (NW * BM)  // 64 query rows per block
#define BN       32         // key block
#define LN_EPS   1e-5f
#define BND      ((size_t)NBATCH * NSEQ * DIMD)   // elements per tensor

// LDS (halfs): Qs[64][256] | K 2x[32][256] | Vt 2x[256][32] | Ps[4][16][32]
#define SMEM_HALFS (BLOCK_M*DIMD + 2*BN*DIMD + 2*DIMD*BN + NW*16*BN)
#define SMEM_BYTES (SMEM_HALFS * 2)

#if __has_builtin(__builtin_amdgcn_global_load_async_to_lds_b128)
#define HAVE_ASYNC 1
#else
#define HAVE_ASYNC 0
#endif

#define AS1 __attribute__((address_space(1)))
#define AS3 __attribute__((address_space(3)))

// explicit LDS (addrspace 3) accessors: guarantee ds_load/ds_store lowering
__device__ __forceinline__ v8h lds_ld8(const f16* p) {
  return *(const AS3 v8h*)(uint32_t)(uintptr_t)p;
}
__device__ __forceinline__ void lds_st_h(f16* p, f16 x) {
  *(AS3 f16*)(uint32_t)(uintptr_t)p = x;
}

// copy 16 bytes global -> LDS (async DMA when available)
__device__ __forceinline__ void cp16(const f16* g, f16* l) {
#if HAVE_ASYNC
  __builtin_amdgcn_global_load_async_to_lds_b128(
      (AS1 v4i*)(uintptr_t)g,
      (AS3 v4i*)(uint32_t)(uintptr_t)l, 0, 0);
#else
  *(AS3 v8h*)(uint32_t)(uintptr_t)l = *(const v8h*)g;
#endif
}
__device__ __forceinline__ void wait_async() {
#if HAVE_ASYNC
  asm volatile("s_wait_asynccnt 0x0" ::: "memory");
#endif
}

__device__ __forceinline__ v16h cat8(v8h a, v8h b) {
  return __builtin_shufflevector(a, b, 0,1,2,3,4,5,6,7,8,9,10,11,12,13,14,15);
}

// ---------------- pass 1a: q,k fp32 -> f16 row-major ----------------
__global__ __launch_bounds__(256)
void cvt_qk(const float* __restrict__ q, const float* __restrict__ k,
            f16* __restrict__ qh, f16* __restrict__ kh)
{
  size_t i = ((size_t)blockIdx.x * 256 + threadIdx.x) * 4;
  v4f xq = *(const v4f*)(q + i);
  v4f xk = *(const v4f*)(k + i);
  v4h hq = { (f16)xq.x, (f16)xq.y, (f16)xq.z, (f16)xq.w };
  v4h hk = { (f16)xk.x, (f16)xk.y, (f16)xk.z, (f16)xk.w };
  *(v4h*)(qh + i) = hq;
  *(v4h*)(kh + i) = hk;
}

// ---------------- pass 1b: v fp32 [b][n][d] -> f16 [b][d][n] ----------------
__global__ __launch_bounds__(256)
void cvt_vt(const float* __restrict__ v, f16* __restrict__ vt)
{
  __shared__ f16 tile[32][33];
  const int b  = blockIdx.z;
  const int n0 = blockIdx.x * 32;
  const int d0 = blockIdx.y * 32;
  const int j  = threadIdx.x & 31;
  const int i  = threadIdx.x >> 5;    // 0..7
  const float* vb = v + ((size_t)b * NSEQ + n0) * DIMD + d0;
  #pragma unroll
  for (int kk = 0; kk < 4; ++kk)
    tile[i + 8 * kk][j] = (f16)vb[(size_t)(i + 8 * kk) * DIMD + j];
  __syncthreads();
  f16* vtb = vt + ((size_t)b * DIMD + d0) * NSEQ + n0;
  #pragma unroll
  for (int kk = 0; kk < 4; ++kk)
    vtb[(size_t)(i + 8 * kk) * NSEQ + j] = tile[j][i + 8 * kk];
}

// ---------------- pass 2: fused flash attention + LayerNorm ----------------
__global__ __launch_bounds__(NW * 32, 1)
void attn_ln_fused(const float* __restrict__ q, const float* __restrict__ gamma,
                   const float* __restrict__ beta, float* __restrict__ out,
                   const f16* __restrict__ qh, const f16* __restrict__ kh,
                   const f16* __restrict__ vt)
{
  extern __shared__ char smem_raw[];
  f16* Qs    = (f16*)smem_raw;                 // [64][256]
  f16* Kbase = Qs + BLOCK_M * DIMD;            // 2x [32][256]
  f16* Vbase = Kbase + 2 * BN * DIMD;          // 2x [256][32] (d-major)
  f16* Ps    = Vbase + 2 * DIMD * BN;          // [NW][16][32]

  const int tid  = threadIdx.x;
  const int wave = tid >> 5;
  const int lane = tid & 31;
  const int half = lane >> 4;
  const int ln   = lane & 15;

  const int blocks_per_batch = NSEQ / BLOCK_M;     // 64
  const int b    = blockIdx.x / blocks_per_batch;
  const int row0 = (blockIdx.x % blocks_per_batch) * BLOCK_M;

  const float* qb  = q  + (size_t)b * NSEQ * DIMD;
  const f16*   qhb = qh + (size_t)b * NSEQ * DIMD;
  const f16*   khb = kh + (size_t)b * NSEQ * DIMD;
  const f16*   vtb = vt + (size_t)b * DIMD * NSEQ;
  float*       ob  = out + (size_t)b * NSEQ * DIMD;

  // ---- issue async staging: Q tile (32KB) + first K/V block ----
  #pragma unroll
  for (int it = 0; it < 16; ++it) {
    int e = (it * (NW * 32) + tid) * 8;            // halfs
    cp16(qhb + (size_t)row0 * DIMD + e, Qs + e);
  }
  #pragma unroll
  for (int it = 0; it < 8; ++it) {                 // K block 0 (contiguous 16KB)
    int e = (it * (NW * 32) + tid) * 8;
    cp16(khb + e, Kbase + e);
  }
  #pragma unroll
  for (int it = 0; it < 8; ++it) {                 // V block 0 (256 rows x 64B)
    int row = it * 32 + (tid >> 2);
    int seg = (tid & 3) * 8;                       // halfs
    cp16(vtb + (size_t)row * NSEQ + seg, Vbase + row * BN + seg);
  }

  // ---- per-wave flash state ----
  v8f o[16];
  #pragma unroll
  for (int t = 0; t < 16; ++t) o[t] = (v8f){0,0,0,0,0,0,0,0};
  float mrow[8], lrow[8];
  #pragma unroll
  for (int r = 0; r < 8; ++r) { mrow[r] = -1e30f; lrow[r] = 0.0f; }
  const float scl2 = 0.0625f * 1.44269504089f;     // (1/sqrt(D)) * log2(e)

  f16* Pw = Ps + wave * 16 * BN;

  for (int ib = 0; ib < NSEQ / BN; ++ib) {
    const int cur = ib & 1;
    wait_async();                // this wave's staged data has landed in LDS
    __syncthreads();             // everyone's data landed; prev compute done

    // issue next block's staging into the other buffer (overlaps compute)
    if (ib + 1 < NSEQ / BN) {
      const int key0n = (ib + 1) * BN;
      f16* Kn = Kbase + (cur ^ 1) * (BN * DIMD);
      f16* Vn = Vbase + (cur ^ 1) * (DIMD * BN);
      #pragma unroll
      for (int it = 0; it < 8; ++it) {
        int e = (it * (NW * 32) + tid) * 8;
        cp16(khb + (size_t)key0n * DIMD + e, Kn + e);
      }
      #pragma unroll
      for (int it = 0; it < 8; ++it) {
        int row = it * 32 + (tid >> 2);
        int seg = (tid & 3) * 8;
        cp16(vtb + (size_t)row * NSEQ + key0n + seg, Vn + row * BN + seg);
      }
    }

    const f16* Ks = Kbase + cur * (BN * DIMD);
    const f16* Vt = Vbase + cur * (DIMD * BN);

    // ---- S = Q K^T : 2 tiles x 8 k-steps of 32 ----
    v8f s[2];
    #pragma unroll
    for (int st = 0; st < 2; ++st) {
      v8f acc = (v8f){0,0,0,0,0,0,0,0};
      #pragma unroll
      for (int ds = 0; ds < 8; ++ds) {
        const f16* qp = Qs + (wave * BM + ln) * DIMD + ds * 32 + half * 8;
        v16h A = cat8(lds_ld8(qp), lds_ld8(qp + 16));
        const f16* kp = Ks + (st * 16 + ln) * DIMD + ds * 32 + half * 8;
        v16h B = cat8(lds_ld8(kp), lds_ld8(kp + 16));
        acc = __builtin_amdgcn_wmma_f32_16x16x32_f16(
                  false, A, false, B, (short)0, acc, false, false);
      }
      s[st] = acc;
    }

    // ---- online softmax in base-2 domain (rows live in one 16-lane half) ----
    float alpha[8];
    #pragma unroll
    for (int r = 0; r < 8; ++r) {
      float mx = fmaxf(s[0][r], s[1][r]);
      mx = fmaxf(mx, __shfl_xor(mx, 1));
      mx = fmaxf(mx, __shfl_xor(mx, 2));
      mx = fmaxf(mx, __shfl_xor(mx, 4));
      mx = fmaxf(mx, __shfl_xor(mx, 8));
      float mnew = fmaxf(mrow[r], mx * scl2);
      alpha[r] = exp2f(mrow[r] - mnew);
      mrow[r] = mnew;
    }
    #pragma unroll
    for (int r = 0; r < 8; ++r) {
      float p0 = exp2f(s[0][r] * scl2 - mrow[r]);
      float p1 = exp2f(s[1][r] * scl2 - mrow[r]);
      f16* pp = Pw + (r + 8 * half) * BN;
      lds_st_h(pp + ln,      (f16)p0);
      lds_st_h(pp + 16 + ln, (f16)p1);
      float rs = p0 + p1;
      rs += __shfl_xor(rs, 1);
      rs += __shfl_xor(rs, 2);
      rs += __shfl_xor(rs, 4);
      rs += __shfl_xor(rs, 8);
      lrow[r] = lrow[r] * alpha[r] + rs;
      #pragma unroll
      for (int t = 0; t < 16; ++t) o[t][r] *= alpha[r];
    }

    // ---- O += P V ----
    {
      const f16* pb = Pw + ln * BN + half * 8;
      v16h A = cat8(lds_ld8(pb), lds_ld8(pb + 16));
      #pragma unroll
      for (int t = 0; t < 16; ++t) {
        const f16* vp = Vt + (t * 16 + ln) * BN + half * 8;
        v16h B = cat8(lds_ld8(vp), lds_ld8(vp + 16));
        o[t] = __builtin_amdgcn_wmma_f32_16x16x32_f16(
                   false, A, false, B, (short)0, o[t], false, false);
      }
    }
  }

  // ---- epilogue: O/l + fp32 residual, LayerNorm over D=256 ----
  float invl[8];
  #pragma unroll
  for (int r = 0; r < 8; ++r) invl[r] = 1.0f / lrow[r];

  const int myrow0 = row0 + wave * BM;
  #pragma unroll
  for (int t = 0; t < 16; ++t) {
    #pragma unroll
    for (int r = 0; r < 8; ++r) {
      float qv = qb[(size_t)(myrow0 + r + 8 * half) * DIMD + t * 16 + ln];
      o[t][r] = o[t][r] * invl[r] + qv;
    }
  }

  float mu[8], rstd[8];
  #pragma unroll
  for (int r = 0; r < 8; ++r) {
    float ssum = 0.0f;
    #pragma unroll
    for (int t = 0; t < 16; ++t) ssum += o[t][r];
    ssum += __shfl_xor(ssum, 1);
    ssum += __shfl_xor(ssum, 2);
    ssum += __shfl_xor(ssum, 4);
    ssum += __shfl_xor(ssum, 8);
    mu[r] = ssum * (1.0f / 256.0f);
  }
  #pragma unroll
  for (int r = 0; r < 8; ++r) {
    float vsum = 0.0f;
    #pragma unroll
    for (int t = 0; t < 16; ++t) {
      float d = o[t][r] - mu[r];
      vsum += d * d;
    }
    vsum += __shfl_xor(vsum, 1);
    vsum += __shfl_xor(vsum, 2);
    vsum += __shfl_xor(vsum, 4);
    vsum += __shfl_xor(vsum, 8);
    rstd[r] = rsqrtf(vsum * (1.0f / 256.0f) + LN_EPS);
  }

  #pragma unroll
  for (int t = 0; t < 16; ++t) {
    float g  = gamma[t * 16 + ln];
    float be = beta[t * 16 + ln];
    #pragma unroll
    for (int r = 0; r < 8; ++r) {
      ob[(size_t)(myrow0 + r + 8 * half) * DIMD + t * 16 + ln] =
          (o[t][r] - mu[r]) * rstd[r] * g + be;
    }
  }
}

extern "C" void kernel_launch(void* const* d_in, const int* in_sizes, int n_in,
                              void* d_out, int out_size, void* d_ws, size_t ws_size,
                              hipStream_t stream) {
  const float* q     = (const float*)d_in[0];
  const float* k     = (const float*)d_in[1];
  const float* v     = (const float*)d_in[2];
  const float* gamma = (const float*)d_in[3];
  const float* beta  = (const float*)d_in[4];
  float* out = (float*)d_out;

  (void)in_sizes; (void)n_in; (void)out_size; (void)ws_size;

  f16* qh = (f16*)d_ws;            // [B][N][D] f16
  f16* kh = qh + BND;              // [B][N][D] f16
  f16* vt = kh + BND;              // [B][D][N] f16 (transposed)

  // pass 1: precision/layout conversion (bandwidth bound)
  cvt_qk<<<dim3((unsigned)(BND / 4 / 256)), dim3(256), 0, stream>>>(q, k, qh, kh);
  cvt_vt<<<dim3(NSEQ / 32, DIMD / 32, NBATCH), dim3(256), 0, stream>>>(v, vt);

  // pass 2: fused attention + LN
  (void)hipFuncSetAttribute((const void*)attn_ln_fused,
                            hipFuncAttributeMaxDynamicSharedMemorySize,
                            SMEM_BYTES);
  dim3 grid(NBATCH * (NSEQ / BLOCK_M));   // 512 blocks
  dim3 block(NW * 32);                    // 4 waves (wave32)
  attn_ln_fused<<<grid, block, SMEM_BYTES, stream>>>(q, gamma, beta, out,
                                                     qh, kh, vt);
}
